// TransformerMapper_43035572306508
// MI455X (gfx1250) — matrix operand order repaired and checked by reading, not compiled, not run on hardware
//
#include <hip/hip_runtime.h>
#include <hip/hip_bf16.h>
#include <math.h>

typedef __attribute__((ext_vector_type(16))) _Float16 v16h;
typedef __attribute__((ext_vector_type(8)))  float    v8f;

#define Dm   768
#define Hh   12
#define DHh  64
#define SEQ  64
#define BATCH 16
#define MEMN 16384
#define TOPK 32
#define DFFm 3072

#if defined(__HIP_DEVICE_COMPILE__) && __has_builtin(__builtin_amdgcn_global_load_async_to_lds_b32)
#define USE_ASYNC_LDS 1
#else
#define USE_ASYNC_LDS 0
#endif

typedef __attribute__((address_space(1))) int* gint_p;
typedef __attribute__((address_space(3))) int* lint_p;

__device__ __forceinline__ void async_copy_b32(const float* g, float* l) {
#if USE_ASYNC_LDS
  __builtin_amdgcn_global_load_async_to_lds_b32(
      (gint_p)(int*)(g), (lint_p)(int*)(l), 0, 0);
#else
  *l = *g;                      // synchronous fallback (still LDS-staged)
#endif
}

__device__ __forceinline__ void wait_async_lds() {
#if USE_ASYNC_LDS
#if __has_builtin(__builtin_amdgcn_s_wait_asynccnt)
  __builtin_amdgcn_s_wait_asynccnt(0);
#else
  asm volatile("s_wait_asynccnt 0x0" ::: "memory");
#endif
#endif
}

__device__ __forceinline__ float gelu_tanh(float x) {
  float x3 = x * x * x;
  return 0.5f * x * (1.0f + tanhf(0.7978845608028654f * (x + 0.044715f * x3)));
}

// ---------------------------------------------------------------------------
// Generic WMMA GEMM: C[M,N] = epilogue(A[M,K] @ B[K,N])
// Block = 128 threads = 4 waves; block tile 64x64; each wave owns 16 rows x 64
// cols (4 accumulators, A fragment reused). Grid = (N/64, ceil(M/64)).
// B tile (32x64) is staged in LDS once per block via async global->LDS b32
// copies (double-buffered, stride-65 padding => bank-conflict-free reads).
// TRANSB: B stored [N,K] row-major (used for q @ mem_k^T scoring).
// ---------------------------------------------------------------------------
template<bool BIAS, bool RES, bool GELU, bool TRANSB>
__global__ void gemm_wmma(const float* __restrict__ A, const float* __restrict__ B,
                          const float* __restrict__ bias, const float* __restrict__ res,
                          float* __restrict__ C, int Mdim, int Ndim, int Kdim) {
  __shared__ float svB[2][32][65];
  const int tid  = threadIdx.x;
  const int lane = tid & 31;
  const int wave = tid >> 5;
  const int row0 = blockIdx.y * 64 + wave * 16;
  const int col0 = blockIdx.x * 64;
  const bool active = (row0 + 16 <= Mdim);   // wave-uniform; EXEC stays full
  const int lm  = lane & 15;
  const int hiA = (lane & 16) ? 8  : 0;      // A: 16-bit 16x32 fragment layout
  const int hiB = (lane & 16) ? 8  : 0;      // B: 32x16, same k order as A (lanes 16-31 hold K=8..15,24..31)

  // Stage one 32x64 B tile (k0..k0+32) into svB[bsel]; 2048 b32 copies, 16/thread.
  auto stage = [&](int k0, int bsel) {
#pragma unroll
    for (int it = 0; it < 16; ++it) {
      const int idx = tid + it * 128;        // 0..2047
      int k, c;
      const float* src;
      if (TRANSB) {
        k = idx & 31; c = idx >> 5;          // lanes walk K => coalesced
        src = B + (size_t)(col0 + c) * Kdim + (k0 + k);
      } else {
        c = idx & 63; k = idx >> 6;          // lanes walk N => coalesced
        src = B + (size_t)(k0 + k) * Ndim + (col0 + c);
      }
      async_copy_b32(src, &svB[bsel][k][c]);
    }
  };

  stage(0, 0);

  v8f acc0 = {}, acc1 = {}, acc2 = {}, acc3 = {};
  for (int k0 = 0; k0 < Kdim; k0 += 32) {
    const int bsel = (k0 >> 5) & 1;
    wait_async_lds();
    __syncthreads();                         // tile bsel published to all waves
    if (k0 + 32 < Kdim) stage(k0 + 32, bsel ^ 1);

    if (active) {
      v16h a;
      const float* Arow = A + (size_t)(row0 + lm) * Kdim + (k0 + hiA);
#pragma unroll
      for (int j = 0; j < 8; ++j)  a[j] = (_Float16)Arow[j];
#pragma unroll
      for (int j = 8; j < 16; ++j) a[j] = (_Float16)Arow[j + 8];

      v16h b0, b1, b2, b3;
#pragma unroll
      for (int j = 0; j < 16; ++j) {
        const float* Bk = &svB[bsel][hiB + j + (j >= 8 ? 8 : 0)][0];
        b0[j] = (_Float16)Bk[lm];
        b1[j] = (_Float16)Bk[lm + 16];
        b2[j] = (_Float16)Bk[lm + 32];
        b3[j] = (_Float16)Bk[lm + 48];
      }
      acc0 = __builtin_amdgcn_wmma_f32_16x16x32_f16(false, a, false, b0, (short)0, acc0, false, false);
      acc1 = __builtin_amdgcn_wmma_f32_16x16x32_f16(false, a, false, b1, (short)0, acc1, false, false);
      acc2 = __builtin_amdgcn_wmma_f32_16x16x32_f16(false, a, false, b2, (short)0, acc2, false, false);
      acc3 = __builtin_amdgcn_wmma_f32_16x16x32_f16(false, a, false, b3, (short)0, acc3, false, false);
    }
    // No trailing barrier needed: next iteration's wait+barrier orders buffer reuse
    // (a wave passes that barrier only after its WMMAs consumed this tile's ds data).
  }

  if (active) {
    const int mrow = (lane & 16) ? 8 : 0;
#pragma unroll
    for (int t = 0; t < 4; ++t) {
      v8f acc = (t == 0) ? acc0 : (t == 1) ? acc1 : (t == 2) ? acc2 : acc3;
      const int c = col0 + t * 16 + lm;
#pragma unroll
      for (int v = 0; v < 8; ++v) {
        const int r = row0 + mrow + v;
        float x = acc[v];
        if (BIAS) x += bias[c];
        if (GELU) x = gelu_tanh(x);
        if (RES)  x += res[(size_t)r * Ndim + c];
        C[(size_t)r * Ndim + c] = x;
      }
    }
  }
}

// ---------------------------------------------------------------------------
// LayerNorm over D=768; one 256-thread block per row.
// ---------------------------------------------------------------------------
__global__ void layernorm_k(const float* __restrict__ src, const float* __restrict__ sc,
                            const float* __restrict__ bi, float* __restrict__ dst) {
  const int row = blockIdx.x;
  const int tid = threadIdx.x;
  const float* x = src + (size_t)row * Dm;
  float v0 = x[tid], v1 = x[tid + 256], v2 = x[tid + 512];
  __shared__ float ssum[256], ssq[256];
  ssum[tid] = v0 + v1 + v2;
  ssq[tid]  = v0 * v0 + v1 * v1 + v2 * v2;
  __syncthreads();
  for (int off = 128; off > 0; off >>= 1) {
    if (tid < off) { ssum[tid] += ssum[tid + off]; ssq[tid] += ssq[tid + off]; }
    __syncthreads();
  }
  const float mean = ssum[0] * (1.0f / Dm);
  const float var  = ssq[0] * (1.0f / Dm) - mean * mean;
  const float rstd = rsqrtf(var + 1e-5f);
  float* y = dst + (size_t)row * Dm;
  y[tid]       = (v0 - mean) * rstd * sc[tid]       + bi[tid];
  y[tid + 256] = (v1 - mean) * rstd * sc[tid + 256] + bi[tid + 256];
  y[tid + 512] = (v2 - mean) * rstd * sc[tid + 512] + bi[tid + 512];
}

// Final LN: only prefix rows (n>=32) are emitted. grid = 16*32 = 512 blocks.
__global__ void final_ln_k(const float* __restrict__ src, const float* __restrict__ sc,
                           const float* __restrict__ bi, float* __restrict__ out) {
  const int ob = blockIdx.x;                // output row 0..511
  const int b  = ob >> 5;
  const int n  = 32 + (ob & 31);
  const int row = b * SEQ + n;
  const int tid = threadIdx.x;
  const float* x = src + (size_t)row * Dm;
  float v0 = x[tid], v1 = x[tid + 256], v2 = x[tid + 512];
  __shared__ float ssum[256], ssq[256];
  ssum[tid] = v0 + v1 + v2;
  ssq[tid]  = v0 * v0 + v1 * v1 + v2 * v2;
  __syncthreads();
  for (int off = 128; off > 0; off >>= 1) {
    if (tid < off) { ssum[tid] += ssum[tid + off]; ssq[tid] += ssq[tid + off]; }
    __syncthreads();
  }
  const float mean = ssum[0] * (1.0f / Dm);
  const float var  = ssq[0] * (1.0f / Dm) - mean * mean;
  const float rstd = rsqrtf(var + 1e-5f);
  float* y = out + (size_t)ob * Dm;
  y[tid]       = (v0 - mean) * rstd * sc[tid]       + bi[tid];
  y[tid + 256] = (v1 - mean) * rstd * sc[tid + 256] + bi[tid + 256];
  y[tid + 512] = (v2 - mean) * rstd * sc[tid + 512] + bi[tid + 512];
}

// hstate[b, n, :] = (n<32) ? t[b, n, :] : prefix[n-32, :]
__global__ void pack_hstate_k(const float* __restrict__ t, const float* __restrict__ pref,
                              float* __restrict__ hstate) {
  const int idx = blockIdx.x * 256 + threadIdx.x;   // < 1024*768
  const int d   = idx % Dm;
  const int tok = idx / Dm;
  const int b = tok >> 6, n = tok & 63;
  hstate[idx] = (n < 32) ? t[(size_t)b * (32 * Dm) + (size_t)n * Dm + d]
                         : pref[(size_t)(n - 32) * Dm + d];
}

// qb[h*64+i, d] = q[b,i, h*64+d] * DH^-0.5   (row-major 768x64 for scoring GEMM)
__global__ void qrepack_k(const float* __restrict__ q, float* __restrict__ qb, int b) {
  const int idx = blockIdx.x * 256 + threadIdx.x;   // < 768*64
  const int d = idx & 63;
  const int r = idx >> 6;                           // h*64+i
  const int h = r >> 6, i = r & 63;
  qb[idx] = q[(size_t)(b * SEQ + i) * Dm + h * DHh + d] * 0.125f;
}

// Top-32 of 16384 scores per row; 32 passes of block-wide argmax.
__global__ void topk_k(const float* __restrict__ scores, float* __restrict__ topv,
                       int* __restrict__ topi, int b) {
  const int row = blockIdx.x;                       // 0..767 (h*64+i)
  const int tid = threadIdx.x;                      // 256
  const float* S = scores + (size_t)row * MEMN;
  float vals[64];
#pragma unroll
  for (int e = 0; e < 64; ++e) vals[e] = S[e * 256 + tid];
  __shared__ float sv[256];
  __shared__ int   si[256];
  const int outbase = ((b * 768) + row) * TOPK;
  for (int k = 0; k < TOPK; ++k) {
    float best = -3.0e38f; int bi = 0;
#pragma unroll
    for (int e = 0; e < 64; ++e)
      if (vals[e] > best) { best = vals[e]; bi = e * 256 + tid; }
    sv[tid] = best; si[tid] = bi;
    __syncthreads();
    for (int off = 128; off > 0; off >>= 1) {
      if (tid < off && sv[tid + off] > sv[tid]) { sv[tid] = sv[tid + off]; si[tid] = si[tid + off]; }
      __syncthreads();
    }
    const float gv = sv[0]; const int gi = si[0];
    if (tid == 0) { topv[outbase + k] = gv; topi[outbase + k] = gi; }
    if ((gi & 255) == tid) vals[gi >> 8] = -3.0e38f;   // owner invalidates
    __syncthreads();
  }
}

// Local causal attention; one block (64 threads) per (b, head).
__global__ void attn_local_k(const float* __restrict__ q, const float* __restrict__ kv,
                             float* __restrict__ ao) {
  const int bh = blockIdx.x;
  const int b = bh / Hh, h = bh % Hh;
  __shared__ float sk[SEQ][DHh];
  __shared__ float svv[SEQ][DHh];
  const int tid = threadIdx.x;
  for (int idx = tid; idx < SEQ * 128; idx += 64) {
    const int j = idx >> 7, d = idx & 127;
    const float val = kv[(size_t)(b * SEQ + j) * 128 + d];
    if (d < 64) sk[j][d] = val; else svv[j][d - 64] = val;
  }
  __syncthreads();
  const int i = tid;
  float qr[DHh];
#pragma unroll
  for (int d = 0; d < DHh; ++d)
    qr[d] = q[(size_t)(b * SEQ + i) * Dm + h * DHh + d] * 0.125f;
  float sim[SEQ];
  float mx = -3.0e38f;
  for (int j = 0; j <= i; ++j) {
    float s = 0.f;
#pragma unroll
    for (int d = 0; d < DHh; ++d) s += qr[d] * sk[j][d];
    sim[j] = s; mx = fmaxf(mx, s);
  }
  float den = 0.f;
  for (int j = 0; j <= i; ++j) { sim[j] = __expf(sim[j] - mx); den += sim[j]; }
  const float inv = 1.0f / den;
  float o[DHh];
#pragma unroll
  for (int d = 0; d < DHh; ++d) o[d] = 0.f;
  for (int j = 0; j <= i; ++j) {
    const float w = sim[j] * inv;
#pragma unroll
    for (int d = 0; d < DHh; ++d) o[d] += w * svv[j][d];
  }
  float* dst = ao + (size_t)(b * SEQ + i) * Dm + h * DHh;
#pragma unroll
  for (int d = 0; d < DHh; ++d) dst[d] = o[d];
}

// Memory-augmented attention: logits = [top_vals(32), causal sim(64)].
__global__ void attn_mem_k(const float* __restrict__ q, const float* __restrict__ kv,
                           const float* __restrict__ topv, const int* __restrict__ topi,
                           const float* __restrict__ mem_v, float* __restrict__ ao) {
  const int bh = blockIdx.x;
  const int b = bh / Hh, h = bh % Hh;
  __shared__ float sk[SEQ][DHh];
  __shared__ float svv[SEQ][DHh];
  const int tid = threadIdx.x;
  for (int idx = tid; idx < SEQ * 128; idx += 64) {
    const int j = idx >> 7, d = idx & 127;
    const float val = kv[(size_t)(b * SEQ + j) * 128 + d];
    if (d < 64) sk[j][d] = val; else svv[j][d - 64] = val;
  }
  __syncthreads();
  const int i = tid;
  float qr[DHh];
#pragma unroll
  for (int d = 0; d < DHh; ++d)
    qr[d] = q[(size_t)(b * SEQ + i) * Dm + h * DHh + d] * 0.125f;

  const int tb = ((b * Hh + h) * SEQ + i) * TOPK;
  float tv[TOPK];
  float mx = -3.0e38f;
#pragma unroll
  for (int k = 0; k < TOPK; ++k) { tv[k] = topv[tb + k]; mx = fmaxf(mx, tv[k]); }

  float sim[SEQ];
  for (int j = 0; j <= i; ++j) {
    float s = 0.f;
#pragma unroll
    for (int d = 0; d < DHh; ++d) s += qr[d] * sk[j][d];
    sim[j] = s; mx = fmaxf(mx, s);
  }
  float den = 0.f;
#pragma unroll
  for (int k = 0; k < TOPK; ++k) { tv[k] = __expf(tv[k] - mx); den += tv[k]; }
  for (int j = 0; j <= i; ++j) { sim[j] = __expf(sim[j] - mx); den += sim[j]; }
  const float inv = 1.0f / den;

  float o[DHh];
#pragma unroll
  for (int d = 0; d < DHh; ++d) o[d] = 0.f;
  for (int j = 0; j <= i; ++j) {
    const float w = sim[j] * inv;
#pragma unroll
    for (int d = 0; d < DHh; ++d) o[d] += w * svv[j][d];
  }
#pragma unroll
  for (int k = 0; k < TOPK; ++k) {
    const float w = tv[k] * inv;
    const float* mv = mem_v + (size_t)(b * MEMN + topi[tb + k]) * DHh;
    for (int d = 0; d < DHh; ++d) o[d] += w * mv[d];
  }
  float* dst = ao + (size_t)(b * SEQ + i) * Dm + h * DHh;
#pragma unroll
  for (int d = 0; d < DHh; ++d) dst[d] = o[d];
}

// ---------------------------------------------------------------------------
extern "C" void kernel_launch(void* const* d_in, const int* in_sizes, int n_in,
                              void* d_out, int out_size, void* d_ws, size_t ws_size,
                              hipStream_t stream) {
  (void)in_sizes; (void)n_in; (void)out_size; (void)ws_size;
  const float* x      = (const float*)d_in[0];
  const float* lin_w  = (const float*)d_in[1];
  const float* lin_b  = (const float*)d_in[2];
  const float* pref   = (const float*)d_in[3];
  const float* ln1_s  = (const float*)d_in[4];
  const float* ln1_b  = (const float*)d_in[5];
  const float* wq     = (const float*)d_in[6];
  const float* wkv    = (const float*)d_in[7];
  const float* wo     = (const float*)d_in[8];
  const float* ln2_s  = (const float*)d_in[9];
  const float* ln2_b  = (const float*)d_in[10];
  const float* w1     = (const float*)d_in[11];
  const float* b1     = (const float*)d_in[12];
  const float* w2     = (const float*)d_in[13];
  const float* b2     = (const float*)d_in[14];
  const float* lnf_s  = (const float*)d_in[15];
  const float* lnf_b  = (const float*)d_in[16];
  const float* mem_k  = (const float*)d_in[17];
  const float* mem_v  = (const float*)d_in[18];
  float* out = (float*)d_out;

  float* ws     = (float*)d_ws;
  float* t_buf  = ws;                                    // 16*24576
  float* hstate = t_buf  + (size_t)16 * 24576;           // 1024*768
  float* hn     = hstate + (size_t)1024 * 768;           // 1024*768
  float* qbuf   = hn     + (size_t)1024 * 768;           // 1024*768
  float* kvbuf  = qbuf   + (size_t)1024 * 768;           // 1024*128
  float* aobuf  = kvbuf  + (size_t)1024 * 128;           // 1024*768
  float* ffbuf  = aobuf  + (size_t)1024 * 768;           // 1024*3072
  float* qb     = ffbuf  + (size_t)1024 * 3072;          // 768*64
  float* scores = qb     + (size_t)768 * 64;             // 768*16384 (reused per batch)
  float* topv   = scores + (size_t)768 * 16384;          // 12288*32
  int*   topi   = (int*)(topv + (size_t)12288 * 32);     // 12288*32

  dim3 blk(128);

  // t = x @ lin_w + lin_b   (16 x 24576, K=768)
  gemm_wmma<true, false, false, false><<<dim3(24576 / 64, 1), blk, 0, stream>>>(
      x, lin_w, lin_b, nullptr, t_buf, 16, 24576, 768);
  pack_hstate_k<<<(1024 * 768) / 256, 256, 0, stream>>>(t_buf, pref, hstate);

  for (int l = 0; l < 8; ++l) {
    layernorm_k<<<1024, 256, 0, stream>>>(hstate, ln1_s + l * Dm, ln1_b + l * Dm, hn);
    gemm_wmma<false, false, false, false><<<dim3(768 / 64, 1024 / 64), blk, 0, stream>>>(
        hn, wq + (size_t)l * 768 * 768, nullptr, nullptr, qbuf, 1024, 768, 768);
    gemm_wmma<false, false, false, false><<<dim3(128 / 64, 1024 / 64), blk, 0, stream>>>(
        hn, wkv + (size_t)l * 768 * 128, nullptr, nullptr, kvbuf, 1024, 128, 768);

    if (l == 4) {
      for (int b = 0; b < BATCH; ++b) {
        qrepack_k<<<(768 * 64) / 256, 256, 0, stream>>>(qbuf, qb, b);
        // scores[768,16384] = qb[768,64] @ mem_k[b]^T (mem_k stored [M,64] => TRANSB)
        gemm_wmma<false, false, false, true><<<dim3(MEMN / 64, 768 / 64), blk, 0, stream>>>(
            qb, mem_k + (size_t)b * MEMN * DHh, nullptr, nullptr, scores, 768, MEMN, 64);
        topk_k<<<768, 256, 0, stream>>>(scores, topv, topi, b);
      }
      attn_mem_k<<<BATCH * Hh, 64, 0, stream>>>(qbuf, kvbuf, topv, topi, mem_v, aobuf);
    } else {
      attn_local_k<<<BATCH * Hh, 64, 0, stream>>>(qbuf, kvbuf, aobuf);
    }

    // hstate = hstate + attnout @ wo
    gemm_wmma<false, true, false, false><<<dim3(768 / 64, 1024 / 64), blk, 0, stream>>>(
        aobuf, wo + (size_t)l * 768 * 768, nullptr, hstate, hstate, 1024, 768, 768);

    layernorm_k<<<1024, 256, 0, stream>>>(hstate, ln2_s + l * Dm, ln2_b + l * Dm, hn);
    // ff = gelu(hn @ w1 + b1)
    gemm_wmma<true, false, true, false><<<dim3(DFFm / 64, 1024 / 64), blk, 0, stream>>>(
        hn, w1 + (size_t)l * 768 * DFFm, b1 + (size_t)l * DFFm, nullptr, ffbuf, 1024, DFFm, 768);
    // hstate = hstate + ff @ w2 + b2
    gemm_wmma<true, true, false, false><<<dim3(768 / 64, 1024 / 64), blk, 0, stream>>>(
        ffbuf, w2 + (size_t)l * DFFm * 768, b2 + (size_t)l * Dm, hstate, hstate, 1024, 768, DFFm);
  }

  final_ln_k<<<512, 256, 0, stream>>>(hstate, lnf_s, lnf_b, out);
}
